// Decoder_43808666419609
// MI455X (gfx1250) — matrix operand order repaired and checked by reading, not compiled, not run on hardware
//
#include <hip/hip_runtime.h>
#include <hip/hip_bf16.h>
#include <math.h>

// Dims (fixed by the reference)
#define NB   64
#define TXD  1024
#define NHD  512
#define HD   512
#define ED   300
#define VD   32000
#define G4D  2048          // 4*H
#define BN_EPS 1e-5f

typedef __attribute__((ext_vector_type(16))) __bf16 v16bf;
typedef __attribute__((ext_vector_type(8)))  float  v8f;

union BF16Frag {
  v16bf v;
  unsigned short u[16];
  uint4 q[2];
};

__device__ __forceinline__ unsigned short f2bf(float f) {
  unsigned int b = __float_as_uint(f);
  b += 0x7FFFu + ((b >> 16) & 1u);      // round-to-nearest-even
  return (unsigned short)(b >> 16);
}
__device__ __forceinline__ float sigf(float x) { return 1.0f / (1.0f + expf(-x)); }

// ---------------------------------------------------------------- pre LSTM gates
__global__ void k_pre_gates(const int* __restrict__ yt, const float* __restrict__ embW,
                            const float* __restrict__ c_prev, const float* __restrict__ h0,
                            const float* __restrict__ Wih, const float* __restrict__ Whh,
                            const float* __restrict__ bih, const float* __restrict__ bhh,
                            float* __restrict__ g) {
  int idx = blockIdx.x * blockDim.x + threadIdx.x;   // 64*2048
  int n = idx >> 11;
  int j = idx & (G4D - 1);
  const float* wih = Wih + (size_t)j * (NHD + ED);   // row length 812
  float acc = bih[j] + bhh[j];
  const float* emb = embW + (size_t)yt[n] * ED;
  for (int k = 0; k < ED; ++k)  acc += emb[k] * wih[k];
  const float* cp = c_prev + n * NHD;
  for (int k = 0; k < NHD; ++k) acc += cp[k] * wih[ED + k];
  const float* whh = Whh + (size_t)j * HD;
  const float* hh  = h0 + n * HD;
  for (int k = 0; k < HD; ++k)  acc += hh[k] * whh[k];
  g[idx] = acc;
}

// ---------------------------------------------------------------- LSTM activation
__global__ void k_lstm_act(const float* __restrict__ g, const float* __restrict__ c_in,
                           float* __restrict__ h_out, float* __restrict__ c_out,
                           unsigned short* __restrict__ h_bf) {
  int idx = blockIdx.x * blockDim.x + threadIdx.x;   // 64*512
  int n = idx >> 9;
  int j = idx & (HD - 1);
  const float* gr = g + (size_t)n * G4D;
  float gi = sigf(gr[j]);
  float gf = sigf(gr[HD + j]);
  float gg = tanhf(gr[2 * HD + j]);
  float go = sigf(gr[3 * HD + j]);
  float c = gf * c_in[idx] + gi * gg;
  float h = go * tanhf(c);
  h_out[idx] = h;
  c_out[idx] = c;
  if (h_bf) h_bf[idx] = f2bf(h);
}

// ---------------------------------------------------------------- s-projection (+b1 folded)
__global__ void k_s_proj(const float* __restrict__ s, const float* __restrict__ W1,
                         const float* __restrict__ b1, float* __restrict__ spb) {
  int idx = blockIdx.x * blockDim.x + threadIdx.x;   // 64*512
  int n = idx >> 9;
  int j = idx & (HD - 1);
  const float* w  = W1 + (size_t)j * (NHD + HD) + NHD;  // s-half of att_W1 row
  const float* sr = s + n * HD;
  float acc = b1[j];
  for (int k = 0; k < HD; ++k) acc += sr[k] * w[k];
  spb[idx] = acc;
}

// ---------------------------------------------------------------- W1 encoder-half -> bf16
__global__ void k_convert_w1(const float* __restrict__ W1, unsigned short* __restrict__ W1b) {
  int idx = blockIdx.x * blockDim.x + threadIdx.x;   // 512*512
  int j = idx >> 9;
  int k = idx & 511;
  W1b[idx] = f2bf(W1[(size_t)j * (NHD + HD) + k]);
}

// ---------------------------------------------------------------- async B staging helper
// Copies one 64 KB column-chunk of W1b (bf16, WMMA-fragment layout) into LDS via the
// async path: GLOBAL_LOAD_ASYNC_TO_LDS_B128, tracked by ASYNCcnt.
__device__ __forceinline__ void stage_B_async(const unsigned short* __restrict__ W1b,
                                              unsigned ldsbase, int nc, int tid) {
  // 4096 16-byte chunks, 128 threads -> 32 async loads per thread
  int tid2      = tid >> 1;
  int h8        = (tid & 1) << 3;                     // 16B half of the 32B fragment
  int col_lane  = tid2 & 15;
  int krow_lane = ((tid2 & 16) ? 8 : 0) + (h8 << 1);
  for (int i = 0; i < 32; ++i) {
    int s  = i * 64 + tid2;          // fragment slot = (kt*4 + jj)*32 + lane
    int jj = (s >> 5) & 3;
    int kt = s >> 7;
    int col  = (nc * 4 + jj) * 16 + col_lane;
    int krow = (kt << 5) + krow_lane;
    const unsigned short* gp = W1b + (size_t)col * NHD + krow;
    unsigned dst = ldsbase + (unsigned)((i * 128 + tid) << 4);
    asm volatile("global_load_async_to_lds_b128 %0, %1, off"
                 :: "v"(dst), "v"(gp) : "memory");
  }
}

// ---------------------------------------------------------------- attention scores
// WMMA bf16, A tile staged once in LDS (pre-packed fragments), B double-buffered in LDS
// via async-to-LDS DMA overlapped with compute; relu + w2-dot fused in epilogue.
__global__ void __launch_bounds__(128)
k_att_scores(const float* __restrict__ enc, const unsigned short* __restrict__ W1b,
             const float* __restrict__ spb, const float* __restrict__ w2,
             const float* __restrict__ b2, float* __restrict__ scores) {
  extern __shared__ char smem[];                     // 196608 B: A 4x16KB | B 2x64KB
  const unsigned smbase = (unsigned)(size_t)smem;    // LDS byte offset (ISA: addr[31:0])

  int wave = threadIdx.x >> 5;
  int lane = threadIdx.x & 31;
  int tile = blockIdx.x * 4 + wave;                  // 0..4095 M-tiles of 16 rows
  int m0 = tile << 4;
  int n  = m0 >> 10;
  int t0 = m0 & (TXD - 1);
  int lh = lane & 15;
  int hi = lane >> 4;
  const float* arow = enc + (size_t)n * TXD * NHD + (size_t)(t0 + lh) * NHD;
  int ak = hi ? 8 : 0;                               // A K-offset per half-wave

  char* aW = smem + wave * 16384;                    // this wave's A fragments
  const char* aL = aW + lane * 32;                   // base for constant-offset DS loads

  // ---- stage A: convert f32 -> bf16 once, pre-packed per (kt, lane) fragment
  for (int kt = 0; kt < 16; ++kt) {
    int kb = kt << 5;
    const float4* p0 = (const float4*)(arow + kb + ak);
    float4 a0 = p0[0], a1 = p0[1];
    const float4* p1 = (const float4*)(arow + kb + 16 + ak);
    float4 a2 = p1[0], a3 = p1[1];
    BF16Frag A;
    A.u[0]  = f2bf(a0.x); A.u[1]  = f2bf(a0.y); A.u[2]  = f2bf(a0.z); A.u[3]  = f2bf(a0.w);
    A.u[4]  = f2bf(a1.x); A.u[5]  = f2bf(a1.y); A.u[6]  = f2bf(a1.z); A.u[7]  = f2bf(a1.w);
    A.u[8]  = f2bf(a2.x); A.u[9]  = f2bf(a2.y); A.u[10] = f2bf(a2.z); A.u[11] = f2bf(a2.w);
    A.u[12] = f2bf(a3.x); A.u[13] = f2bf(a3.y); A.u[14] = f2bf(a3.z); A.u[15] = f2bf(a3.w);
    uint4* dp = (uint4*)(aW + kt * 1024 + lane * 32);
    dp[0] = A.q[0];
    dp[1] = A.q[1];
  }

  // ---- prologue: stage first B chunk, wait, publish
  stage_B_async(W1b, smbase + 0x10000u, 0, threadIdx.x);
  asm volatile("s_wait_asynccnt 0x0" ::: "memory");
  __syncthreads();

  float partial[8];
#pragma unroll
  for (int r = 0; r < 8; ++r) partial[r] = 0.0f;

  unsigned cur = 0x10000u;                           // buffer offset holding chunk nc
  for (int nc = 0; nc < 8; ++nc) {                   // 8 chunks x 4 N-tiles = 512 cols
    // prefetch next chunk into the other buffer while we compute on this one
    if (nc < 7) stage_B_async(W1b, smbase + (cur ^ 0x30000u), nc + 1, threadIdx.x);
    const char* bL = smem + cur + lane * 32;         // constant-offset DS base

    v8f acc[4];
#pragma unroll
    for (int jj = 0; jj < 4; ++jj) { v8f zz = {}; acc[jj] = zz; }

    for (int kt = 0; kt < 16; ++kt) {
      BF16Frag A;
      const uint4* ap = (const uint4*)(aL + kt * 1024);
      A.q[0] = ap[0];
      A.q[1] = ap[1];
      BF16Frag B[4];
#pragma unroll
      for (int jj = 0; jj < 4; ++jj) {
        const uint4* bp = (const uint4*)(bL + kt * 4096 + jj * 1024);
        B[jj].q[0] = bp[0];
        B[jj].q[1] = bp[1];
      }
#pragma unroll
      for (int jj = 0; jj < 4; ++jj) {
        acc[jj] = __builtin_amdgcn_wmma_f32_16x16x32_bf16(
            false, A.v, false, B[jj].v, (short)0, acc[jj], false, false);
      }
    }
    // epilogue: relu(enc_proj + s_proj + b1) dotted with w2, per row
#pragma unroll
    for (int jj = 0; jj < 4; ++jj) {
      int col = ((nc * 4 + jj) << 4) + lh;
      float sp = spb[n * HD + col];
      float wv = w2[col];
#pragma unroll
      for (int r = 0; r < 8; ++r) {
        float e = acc[jj][r] + sp;
        e = e > 0.0f ? e : 0.0f;
        partial[r] += wv * e;
      }
    }
    // retire this iteration's prefetch and publish the buffer to all waves
    asm volatile("s_wait_asynccnt 0x0" ::: "memory");
    __syncthreads();
    cur ^= 0x30000u;
  }

  // reduce across the 16 lanes sharing each C-row (lanes 0-15: rows r, 16-31: rows 8+r)
#pragma unroll
  for (int off = 1; off < 16; off <<= 1) {
#pragma unroll
    for (int r = 0; r < 8; ++r) partial[r] += __shfl_xor(partial[r], off, 32);
  }
  float bb2 = b2[0];
  if (lane == 0) {
#pragma unroll
    for (int r = 0; r < 8; ++r) scores[(size_t)n * TXD + t0 + r] = partial[r] + bb2;
  } else if (lane == 16) {
#pragma unroll
    for (int r = 0; r < 8; ++r) scores[(size_t)n * TXD + t0 + 8 + r] = partial[r] + bb2;
  }
}

// ---------------------------------------------------------------- softmax over Tx (in place)
__global__ void k_softmax_tx(float* __restrict__ scores) {
  __shared__ float red[256];
  int n = blockIdx.x, tid = threadIdx.x;
  float* row = scores + (size_t)n * TXD;
  float v[4];
  float m = -1e30f;
#pragma unroll
  for (int i = 0; i < 4; ++i) { v[i] = row[tid + (i << 8)]; m = fmaxf(m, v[i]); }
  red[tid] = m; __syncthreads();
  for (int s = 128; s > 0; s >>= 1) { if (tid < s) red[tid] = fmaxf(red[tid], red[tid + s]); __syncthreads(); }
  m = red[0]; __syncthreads();
  float sum = 0.0f;
#pragma unroll
  for (int i = 0; i < 4; ++i) { v[i] = expf(v[i] - m); sum += v[i]; }
  red[tid] = sum; __syncthreads();
  for (int s = 128; s > 0; s >>= 1) { if (tid < s) red[tid] += red[tid + s]; __syncthreads(); }
  float inv = 1.0f / red[0];
#pragma unroll
  for (int i = 0; i < 4; ++i) row[tid + (i << 8)] = v[i] * inv;
}

// ---------------------------------------------------------------- context = alphas^T * enc (L2-hot pass)
__global__ void k_context(const float* __restrict__ enc, const float* __restrict__ alphas,
                          float* __restrict__ ctx) {
  __shared__ float al[TXD];
  int n = blockIdx.x, c = threadIdx.x;                // 512 threads
  al[c]       = alphas[(size_t)n * TXD + c];
  al[c + 512] = alphas[(size_t)n * TXD + c + 512];
  __syncthreads();
  const float* en = enc + (size_t)n * TXD * NHD + c;
  float acc = 0.0f;
  for (int t = 0; t < TXD; ++t) acc += al[t] * en[(size_t)t * NHD];
  ctx[n * NHD + c] = acc;
}

// ---------------------------------------------------------------- post LSTM gates
__global__ void k_post_gates(const float* __restrict__ s, const float* __restrict__ ctx,
                             const float* __restrict__ h1,
                             const float* __restrict__ Wih, const float* __restrict__ Whh,
                             const float* __restrict__ bih, const float* __restrict__ bhh,
                             float* __restrict__ g) {
  int idx = blockIdx.x * blockDim.x + threadIdx.x;   // 64*2048
  int n = idx >> 11;
  int j = idx & (G4D - 1);
  const float* wih = Wih + (size_t)j * (HD + NHD);   // row length 1024
  float acc = bih[j] + bhh[j];
  const float* sr = s + n * HD;
  for (int k = 0; k < HD; ++k)  acc += sr[k] * wih[k];
  const float* cc = ctx + n * NHD;
  for (int k = 0; k < NHD; ++k) acc += cc[k] * wih[HD + k];
  const float* whh = Whh + (size_t)j * HD;
  const float* hh  = h1 + n * HD;
  for (int k = 0; k < HD; ++k)  acc += hh[k] * whh[k];
  g[idx] = acc;
}

// ---------------------------------------------------------------- classifier (WMMA)
// Each wave owns a 32-column strip and all 4 M-tiles: every converted mlp_W fragment is
// reused by 4 WMMAs, so the 64 MB weight matrix is streamed from HBM exactly once.
__global__ void __launch_bounds__(256)
k_classifier(const unsigned short* __restrict__ h1b, const float* __restrict__ W,
             const float* __restrict__ b, float* __restrict__ z) {
  int wave = threadIdx.x >> 5;
  int lane = threadIdx.x & 31;
  int wg = blockIdx.x * 8 + wave;                    // 0..999 column strips of 32
  int lh = lane & 15;
  int hi = lane >> 4;
  int ak = hi ? 16 : 0;
  int bk = hi ? 16 : 0;

  v8f acc[4][2];
#pragma unroll
  for (int mt = 0; mt < 4; ++mt)
#pragma unroll
    for (int jj = 0; jj < 2; ++jj) { v8f zz = {}; acc[mt][jj] = zz; }

  for (int kt = 0; kt < 16; ++kt) {                  // K = 512
    int kb = kt << 5;
    BF16Frag A[4];
#pragma unroll
    for (int mt = 0; mt < 4; ++mt) {
      const unsigned short* ar = h1b + (size_t)((mt << 4) + lh) * HD + kb + ak;
      A[mt].q[0] = *(const uint4*)ar;
      A[mt].q[1] = *(const uint4*)(ar + 8);
    }
#pragma unroll
    for (int jj = 0; jj < 2; ++jj) {
      int col = (wg * 2 + jj) * 16 + lh;
      const float4* pw = (const float4*)(W + (size_t)col * HD + kb + bk);
      float4 w0 = pw[0], w1 = pw[1], w2 = pw[2], w3 = pw[3];
      BF16Frag B;
      B.u[0]  = f2bf(w0.x); B.u[1]  = f2bf(w0.y); B.u[2]  = f2bf(w0.z); B.u[3]  = f2bf(w0.w);
      B.u[4]  = f2bf(w1.x); B.u[5]  = f2bf(w1.y); B.u[6]  = f2bf(w1.z); B.u[7]  = f2bf(w1.w);
      B.u[8]  = f2bf(w2.x); B.u[9]  = f2bf(w2.y); B.u[10] = f2bf(w2.z); B.u[11] = f2bf(w2.w);
      B.u[12] = f2bf(w3.x); B.u[13] = f2bf(w3.y); B.u[14] = f2bf(w3.z); B.u[15] = f2bf(w3.w);
#pragma unroll
      for (int mt = 0; mt < 4; ++mt) {
        acc[mt][jj] = __builtin_amdgcn_wmma_f32_16x16x32_bf16(
            false, A[mt].v, false, B.v, (short)0, acc[mt][jj], false, false);
      }
    }
  }
#pragma unroll
  for (int mt = 0; mt < 4; ++mt) {
#pragma unroll
    for (int jj = 0; jj < 2; ++jj) {
      int col = (wg * 2 + jj) * 16 + lh;
      float bias = b[col];
#pragma unroll
      for (int r = 0; r < 8; ++r) {
        int row = (mt << 4) + r + (hi ? 8 : 0);
        float val = acc[mt][jj][r] + bias;
        z[(size_t)row * VD + col] = val > 0.0f ? val : 0.0f;
      }
    }
  }
}

// ---------------------------------------------------------------- BN train-mode stats -> scale/shift
__global__ void k_bn_stats(const float* __restrict__ z, const float* __restrict__ gamma,
                           const float* __restrict__ beta, float* __restrict__ scale,
                           float* __restrict__ shift) {
  int v = blockIdx.x * blockDim.x + threadIdx.x;     // 32000
  float s = 0.0f, s2 = 0.0f;
  for (int n = 0; n < NB; ++n) {
    float x = z[(size_t)n * VD + v];
    s += x; s2 += x * x;
  }
  float mean = s * (1.0f / NB);
  float var  = s2 * (1.0f / NB) - mean * mean;       // biased, matches reference
  float sc = gamma[v] * rsqrtf(var + BN_EPS);
  scale[v] = sc;
  shift[v] = beta[v] - mean * sc;
}

// ---------------------------------------------------------------- final softmax over V per row
__global__ void k_out_softmax(const float* __restrict__ z, const float* __restrict__ scale,
                              const float* __restrict__ shift, float* __restrict__ out) {
  __shared__ float red[256];
  int n = blockIdx.x, tid = threadIdx.x;
  const float* zr = z + (size_t)n * VD;
  float m = -1e30f;
  for (int v = tid; v < VD; v += 256) m = fmaxf(m, zr[v] * scale[v] + shift[v]);
  red[tid] = m; __syncthreads();
  for (int s = 128; s > 0; s >>= 1) { if (tid < s) red[tid] = fmaxf(red[tid], red[tid + s]); __syncthreads(); }
  m = red[0]; __syncthreads();
  float sum = 0.0f;
  for (int v = tid; v < VD; v += 256) sum += expf(zr[v] * scale[v] + shift[v] - m);
  red[tid] = sum; __syncthreads();
  for (int s = 128; s > 0; s >>= 1) { if (tid < s) red[tid] += red[tid + s]; __syncthreads(); }
  float inv = 1.0f / red[0];
  float* outr = out + (size_t)n * VD;
  for (int v = tid; v < VD; v += 256) outr[v] = expf(zr[v] * scale[v] + shift[v] - m) * inv;
}

// ---------------------------------------------------------------- launch
extern "C" void kernel_launch(void* const* d_in, const int* in_sizes, int n_in,
                              void* d_out, int out_size, void* d_ws, size_t ws_size,
                              hipStream_t stream) {
  (void)in_sizes; (void)n_in; (void)out_size; (void)ws_size;
  const int*   yt      = (const int*)d_in[0];
  const float* h0      = (const float*)d_in[1];
  const float* c0      = (const float*)d_in[2];
  const float* h1      = (const float*)d_in[3];
  const float* c1      = (const float*)d_in[4];
  const float* enc     = (const float*)d_in[5];
  const float* c_prev  = (const float*)d_in[6];
  const float* embW    = (const float*)d_in[7];
  const float* preWih  = (const float*)d_in[8];
  const float* preWhh  = (const float*)d_in[9];
  const float* prebih  = (const float*)d_in[10];
  const float* prebhh  = (const float*)d_in[11];
  const float* postWih = (const float*)d_in[12];
  const float* postWhh = (const float*)d_in[13];
  const float* postbih = (const float*)d_in[14];
  const float* postbhh = (const float*)d_in[15];
  const float* attW1   = (const float*)d_in[16];
  const float* attb1   = (const float*)d_in[17];
  const float* attW2   = (const float*)d_in[18];
  const float* attb2   = (const float*)d_in[19];
  const float* mlpW    = (const float*)d_in[20];
  const float* mlpb    = (const float*)d_in[21];
  const float* gamma   = (const float*)d_in[22];
  const float* beta    = (const float*)d_in[23];

  float* out     = (float*)d_out;                    // [64,32000]
  float* h0_out  = out + (size_t)NB * VD;            // [64,512]
  float* c0_out  = h0_out + NB * HD;
  float* h1_out  = c0_out + NB * HD;
  float* c1_out  = h1_out + NB * HD;
  float* ctx_out = c1_out + NB * HD;

  char* ws = (char*)d_ws;
  float*          g_pre  = (float*)(ws + 0);             // 512 KB
  float*          g_post = (float*)(ws + 524288);        // 512 KB
  float*          spb    = (float*)(ws + 1048576);       // 128 KB
  float*          scores = (float*)(ws + 1179648);       // 256 KB (scores -> alphas in place)
  unsigned short* W1b    = (unsigned short*)(ws + 1441792);  // 512 KB bf16
  unsigned short* h1b    = (unsigned short*)(ws + 1966080);  // 64 KB bf16
  float*          z      = (float*)(ws + 2031616);       // 8 MB
  float*          bscale = (float*)(ws + 10223616);      // 128 KB
  float*          bshift = (float*)(ws + 10351616);      // 128 KB

  k_convert_w1 <<<1024, 256, 0, stream>>>(attW1, W1b);
  k_pre_gates  <<<512, 256, 0, stream>>>(yt, embW, c_prev, h0,
                                         preWih, preWhh, prebih, prebhh, g_pre);
  k_lstm_act   <<<128, 256, 0, stream>>>(g_pre, c0, h0_out, c0_out, (unsigned short*)nullptr);
  k_s_proj     <<<128, 256, 0, stream>>>(h0_out, attW1, attb1, spb);
  k_att_scores <<<1024, 128, 196608, stream>>>(enc, W1b, spb, attW2, attb2, scores);
  k_softmax_tx <<<64, 256, 0, stream>>>(scores);
  k_context    <<<64, 512, 0, stream>>>(enc, scores, ctx_out);
  k_post_gates <<<512, 256, 0, stream>>>(h0_out, ctx_out, h1,
                                         postWih, postWhh, postbih, postbhh, g_post);
  k_lstm_act   <<<128, 256, 0, stream>>>(g_post, c1, h1_out, c1_out, h1b);
  k_classifier <<<125, 256, 0, stream>>>(h1b, mlpW, mlpb, z);
  k_bn_stats   <<<125, 256, 0, stream>>>(z, gamma, beta, bscale, bshift);
  k_out_softmax<<<64, 256, 0, stream>>>(z, bscale, bshift, out);
}
